// MotionGenerator_40389872452236
// MI455X (gfx1250) — compile-verified, matrix operand order checked by reading, not run on hardware
//
#include <hip/hip_runtime.h>
#include <hip/hip_bf16.h>

// ---------------------------------------------------------------------------
// Types for CDNA5 WMMA (gfx1250, wave32)
// ---------------------------------------------------------------------------
typedef __attribute__((ext_vector_type(16))) __bf16 v16bf;
typedef __attribute__((ext_vector_type(8)))  float  v8f;
typedef __attribute__((ext_vector_type(4)))  unsigned int u32x4;

// A/B fragment: 16 bf16 halves per lane = two 16-byte chunks.
// Layout (ISA 7.12.2, 16-bit A 16x32): lane l -> row (l&15), kb=(l>>4)*8,
// halves 0..7 = K kb..kb+7, halves 8..15 = K kb+16..kb+23.
union Frag {
    u32x4 u[2];
    v16bf v;
};

__device__ __forceinline__ v8f wmma_bf16(v16bf a, v16bf b, v8f c) {
    return __builtin_amdgcn_wmma_f32_16x16x32_bf16(
        /*neg_a=*/false, a, /*neg_b=*/false, b,
        /*c_mod=*/(short)0, c, /*reuse_a=*/false, /*reuse_b=*/false);
}

__device__ __forceinline__ unsigned short f2bf(float f) {
    union { float f; unsigned u; } v; v.f = f;
    unsigned u = v.u;
    u += 0x7FFFu + ((u >> 16) & 1u);   // round-to-nearest-even
    return (unsigned short)(u >> 16);
}

__device__ __forceinline__ float sigm_f(float x) {
    x = fminf(30.f, fmaxf(-30.f, x));
    return 1.f / (1.f + __expf(-x));
}
__device__ __forceinline__ float tanh_f(float x) {
    x = fminf(15.f, fmaxf(-15.f, x));
    float e = __expf(2.f * x);
    return (e - 1.f) / (e + 1.f);
}

// Problem constants
#define BN   512            // batch
#define HN   512            // hidden
#define KXH  1024           // concat [x|h] inner dim
#define G4H  2048           // 4*H gates
#define KIN  736            // 713 padded to multiple of 32
#define KINR 713
#define FN   72
#define FNP  80
#define TT   120

// ---------------------------------------------------------------------------
// Fused LSTM layer-step:
//   gates = [x|h] @ Wcat^T (+ bih + bhh); cell update in registers.
// xh_in  : [512,1024] bf16  (x | h_prev)
// Wcat   : [2048,1024] bf16 (Wih | Whh per gate row)
// c      : [512,512] f32    (in/out, per-element owned by one lane)
// xh_self: write h_new -> xh_self[:,512:1024]  (next step's h input)
// xh_nx  : optional, write h_new -> xh_nx[:,0:512] (next layer's x, this step)
// Grid: 128 blocks x 128 threads (512 waves). Wave = 32 M-rows x 16 N-cols x 4 gates.
// ---------------------------------------------------------------------------
__global__ void lstm_step_kernel(const unsigned short* __restrict__ xh_in,
                                 const unsigned short* __restrict__ Wcat,
                                 const float* __restrict__ bih,
                                 const float* __restrict__ bhh,
                                 float* __restrict__ c,
                                 unsigned short* __restrict__ xh_self,
                                 unsigned short* __restrict__ xh_nx) {
    const int lane = threadIdx.x & 31;
    const int wid  = blockIdx.x * 4 + (threadIdx.x >> 5);  // 0..511
    const int mt   = wid & 15;                              // 16 tiles of 32 rows
    const int nt   = wid >> 4;                              // 32 tiles of 16 cols
    const int m0   = mt * 32;
    const int n0   = nt * 16;

    const int r  = lane & 15;
    const int kb = (lane >> 4) * 8;

    const unsigned short* Ap0 = xh_in + (m0 + r) * KXH + kb;
    const unsigned short* Ap1 = Ap0 + 16 * KXH;
    const unsigned short* Bp0 = Wcat + (0 * HN + n0 + r) * KXH + kb;
    const unsigned short* Bp1 = Wcat + (1 * HN + n0 + r) * KXH + kb;
    const unsigned short* Bp2 = Wcat + (2 * HN + n0 + r) * KXH + kb;
    const unsigned short* Bp3 = Wcat + (3 * HN + n0 + r) * KXH + kb;

    const v8f z = {0.f,0.f,0.f,0.f,0.f,0.f,0.f,0.f};
    v8f acc[2][4];
    #pragma unroll
    for (int h = 0; h < 2; ++h)
        #pragma unroll
        for (int g = 0; g < 4; ++g) acc[h][g] = z;

    for (int kt = 0; kt < KXH; kt += 32) {
        Frag a0, a1;
        a0.u[0] = *(const u32x4*)(Ap0 + kt);
        a0.u[1] = *(const u32x4*)(Ap0 + kt + 16);
        a1.u[0] = *(const u32x4*)(Ap1 + kt);
        a1.u[1] = *(const u32x4*)(Ap1 + kt + 16);
        Frag b;
        b.u[0] = *(const u32x4*)(Bp0 + kt);
        b.u[1] = *(const u32x4*)(Bp0 + kt + 16);
        acc[0][0] = wmma_bf16(a0.v, b.v, acc[0][0]);
        acc[1][0] = wmma_bf16(a1.v, b.v, acc[1][0]);
        b.u[0] = *(const u32x4*)(Bp1 + kt);
        b.u[1] = *(const u32x4*)(Bp1 + kt + 16);
        acc[0][1] = wmma_bf16(a0.v, b.v, acc[0][1]);
        acc[1][1] = wmma_bf16(a1.v, b.v, acc[1][1]);
        b.u[0] = *(const u32x4*)(Bp2 + kt);
        b.u[1] = *(const u32x4*)(Bp2 + kt + 16);
        acc[0][2] = wmma_bf16(a0.v, b.v, acc[0][2]);
        acc[1][2] = wmma_bf16(a1.v, b.v, acc[1][2]);
        b.u[0] = *(const u32x4*)(Bp3 + kt);
        b.u[1] = *(const u32x4*)(Bp3 + kt + 16);
        acc[0][3] = wmma_bf16(a0.v, b.v, acc[0][3]);
        acc[1][3] = wmma_bf16(a1.v, b.v, acc[1][3]);
    }

    // epilogue: C layout -> lane l owns col n0+(l&15), rows rbase..rbase+7 per VGPR
    const int n = n0 + (lane & 15);
    const float bi = bih[0 * HN + n] + bhh[0 * HN + n];
    const float bf = bih[1 * HN + n] + bhh[1 * HN + n];
    const float bg = bih[2 * HN + n] + bhh[2 * HN + n];
    const float bo = bih[3 * HN + n] + bhh[3 * HN + n];
    const int rbase = (lane >> 4) * 8;

    #pragma unroll
    for (int h = 0; h < 2; ++h) {
        #pragma unroll
        for (int rr = 0; rr < 8; ++rr) {
            const int m = m0 + h * 16 + rbase + rr;
            const float gi = sigm_f(acc[h][0][rr] + bi);
            const float gf = sigm_f(acc[h][1][rr] + bf);
            const float gg = tanh_f(acc[h][2][rr] + bg);
            const float go = sigm_f(acc[h][3][rr] + bo);
            const float cold = c[m * HN + n];
            const float cn = gf * cold + gi * gg;
            const float hn = go * tanh_f(cn);
            c[m * HN + n] = cn;
            const unsigned short hb = f2bf(hn);
            xh_self[m * KXH + HN + n] = hb;
            if (xh_nx) xh_nx[m * KXH + n] = hb;
        }
    }
}

// ---------------------------------------------------------------------------
// Decoder embedder GEMM: x = prelu(gen_in[512,736] @ embWp[512,736]^T + b)
// writes bf16 into xh0[:,0:512]. Grid 128x128 (512 waves, 32Mx16N each).
// ---------------------------------------------------------------------------
__global__ void emb_gemm_kernel(const unsigned short* __restrict__ gen_in,
                                const unsigned short* __restrict__ Wp,
                                const float* __restrict__ bias,
                                const float* __restrict__ prelu_a,
                                unsigned short* __restrict__ xh0) {
    const int lane = threadIdx.x & 31;
    const int wid  = blockIdx.x * 4 + (threadIdx.x >> 5);
    const int m0   = (wid & 15) * 32;
    const int n0   = (wid >> 4) * 16;
    const int r  = lane & 15;
    const int kb = (lane >> 4) * 8;

    const unsigned short* Ap0 = gen_in + (m0 + r) * KIN + kb;
    const unsigned short* Ap1 = Ap0 + 16 * KIN;
    const unsigned short* Bp  = Wp + (n0 + r) * KIN + kb;

    const v8f z = {0.f,0.f,0.f,0.f,0.f,0.f,0.f,0.f};
    v8f acc0 = z, acc1 = z;
    for (int kt = 0; kt < KIN; kt += 32) {
        Frag a0, a1, b;
        a0.u[0] = *(const u32x4*)(Ap0 + kt);
        a0.u[1] = *(const u32x4*)(Ap0 + kt + 16);
        a1.u[0] = *(const u32x4*)(Ap1 + kt);
        a1.u[1] = *(const u32x4*)(Ap1 + kt + 16);
        b.u[0]  = *(const u32x4*)(Bp + kt);
        b.u[1]  = *(const u32x4*)(Bp + kt + 16);
        acc0 = wmma_bf16(a0.v, b.v, acc0);
        acc1 = wmma_bf16(a1.v, b.v, acc1);
    }

    const int n = n0 + (lane & 15);
    const float bv = bias[n];
    const float a  = prelu_a[0];
    const int rbase = (lane >> 4) * 8;
    #pragma unroll
    for (int rr = 0; rr < 8; ++rr) {
        const int m0r = m0 + rbase + rr;
        float v0 = acc0[rr] + bv; v0 = v0 >= 0.f ? v0 : a * v0;
        float v1 = acc1[rr] + bv; v1 = v1 >= 0.f ? v1 : a * v1;
        xh0[m0r * KXH + n] = f2bf(v0);
        xh0[(m0r + 16) * KXH + n] = f2bf(v1);
    }
}

// ---------------------------------------------------------------------------
// Output GEMM: out = h[512,512] @ outWp[80,512]^T + b ; store fp32 to d_out
// and bf16 feedback into gen_in[:,0:72]. Grid 20x128 (80 waves).
// ---------------------------------------------------------------------------
__global__ void out_gemm_kernel(const unsigned short* __restrict__ hlast, // xh1 + 512
                                const unsigned short* __restrict__ Wp,
                                const float* __restrict__ bias,
                                int t,
                                float* __restrict__ out,
                                unsigned short* __restrict__ gen_in) {
    const int lane = threadIdx.x & 31;
    const int wid  = blockIdx.x * 4 + (threadIdx.x >> 5);   // 0..79
    const int m0   = (wid % 16) * 32;
    const int n0   = (wid / 16) * 16;                       // 0..64
    const int r  = lane & 15;
    const int kb = (lane >> 4) * 8;

    const unsigned short* Ap0 = hlast + (m0 + r) * KXH + kb;
    const unsigned short* Ap1 = Ap0 + 16 * KXH;
    const unsigned short* Bp  = Wp + (n0 + r) * HN + kb;

    const v8f z = {0.f,0.f,0.f,0.f,0.f,0.f,0.f,0.f};
    v8f acc0 = z, acc1 = z;
    for (int kt = 0; kt < HN; kt += 32) {
        Frag a0, a1, b;
        a0.u[0] = *(const u32x4*)(Ap0 + kt);
        a0.u[1] = *(const u32x4*)(Ap0 + kt + 16);
        a1.u[0] = *(const u32x4*)(Ap1 + kt);
        a1.u[1] = *(const u32x4*)(Ap1 + kt + 16);
        b.u[0]  = *(const u32x4*)(Bp + kt);
        b.u[1]  = *(const u32x4*)(Bp + kt + 16);
        acc0 = wmma_bf16(a0.v, b.v, acc0);
        acc1 = wmma_bf16(a1.v, b.v, acc1);
    }

    const int n = n0 + (lane & 15);
    if (n < FN) {
        const float bv = bias[n];
        const int rbase = (lane >> 4) * 8;
        #pragma unroll
        for (int rr = 0; rr < 8; ++rr) {
            const int ma = m0 + rbase + rr;
            const int mb = ma + 16;
            const float v0 = acc0[rr] + bv;
            const float v1 = acc1[rr] + bv;
            out[(ma * TT + t) * FN + n] = v0;
            out[(mb * TT + t) * FN + n] = v1;
            gen_in[ma * KIN + n] = f2bf(v0);
            gen_in[mb * KIN + n] = f2bf(v1);
        }
    }
}

// ---------------------------------------------------------------------------
// Encoder per-step input: x = prelu(traj[:,t,:3] @ encW^T + b) -> xh0 x-part
// ---------------------------------------------------------------------------
__global__ void traj_embed_kernel(const float* __restrict__ traj,
                                  const float* __restrict__ W,   // [512,3]
                                  const float* __restrict__ b,
                                  const float* __restrict__ prelu_a,
                                  int t,
                                  unsigned short* __restrict__ xh0) {
    const int i = blockIdx.x * blockDim.x + threadIdx.x;   // 0..262143
    const int bi = i >> 9;
    const int j  = i & 511;
    const float* tr = traj + (bi * TT + t) * 3;
    float v = tr[0] * W[j * 3 + 0] + tr[1] * W[j * 3 + 1] + tr[2] * W[j * 3 + 2] + b[j];
    const float a = prelu_a[0];
    v = v >= 0.f ? v : a * v;
    xh0[bi * KXH + j] = f2bf(v);
}

// ---------------------------------------------------------------------------
// Setup kernels
// ---------------------------------------------------------------------------
__global__ void pack_lstm_w_kernel(const float* __restrict__ Wih,
                                   const float* __restrict__ Whh,
                                   unsigned short* __restrict__ Wcat) {
    const int i = blockIdx.x * blockDim.x + threadIdx.x;   // 2048*1024
    const int nrow = i >> 10;
    const int k = i & 1023;
    const float v = (k < HN) ? Wih[nrow * HN + k] : Whh[nrow * HN + (k - HN)];
    Wcat[i] = f2bf(v);
}

__global__ void pack_emb_w_kernel(const float* __restrict__ W,   // [512,713]
                                  unsigned short* __restrict__ Wp) {
    const int i = blockIdx.x * blockDim.x + threadIdx.x;   // 512*736
    if (i >= 512 * KIN) return;
    const int j = i / KIN;
    const int k = i % KIN;
    Wp[i] = (k < KINR) ? f2bf(W[j * KINR + k]) : (unsigned short)0;
}

__global__ void pack_out_w_kernel(const float* __restrict__ W,   // [72,512]
                                  unsigned short* __restrict__ Wp) {
    const int i = blockIdx.x * blockDim.x + threadIdx.x;   // 80*512
    const int j = i >> 9;
    const int k = i & 511;
    Wp[i] = (j < FN) ? f2bf(W[j * HN + k]) : (unsigned short)0;
}

__global__ void build_genin_kernel(const float* __restrict__ label,   // [512,10]
                                   const float* __restrict__ cemb,    // [512,64]
                                   unsigned short* __restrict__ gen_in) {
    const int i = blockIdx.x * blockDim.x + threadIdx.x;   // 512*736
    if (i >= 512 * KIN) return;
    const int bi = i / KIN;
    const int q  = i % KIN;
    float v = 0.f;
    if (q >= FN && q < 712) {
        const int qq = q - FN;
        v = label[bi * 10 + (qq >> 6)] * cemb[bi * 64 + (qq & 63)];
    }
    gen_in[i] = f2bf(v);
}

__global__ void zero_u32_kernel(unsigned int* __restrict__ p, int n) {
    const int i = blockIdx.x * blockDim.x + threadIdx.x;
    if (i < n) p[i] = 0u;
}

__global__ void set_tfrac_kernel(unsigned short* __restrict__ gen_in, float ts) {
    const int b = blockIdx.x * blockDim.x + threadIdx.x;
    if (b < BN) gen_in[b * KIN + 712] = f2bf(ts);
}

// ---------------------------------------------------------------------------
// Host driver
// ---------------------------------------------------------------------------
extern "C" void kernel_launch(void* const* d_in, const int* in_sizes, int n_in,
                              void* d_out, int out_size, void* d_ws, size_t ws_size,
                              hipStream_t stream) {
    const float* label = (const float*)d_in[1];
    const float* cemb  = (const float*)d_in[2];
    const float* traj  = (const float*)d_in[3];
    const float* encW  = (const float*)d_in[4];
    const float* encb  = (const float*)d_in[5];
    const float* prelu = (const float*)d_in[6];
    const float* embW  = (const float*)d_in[7];
    const float* embb  = (const float*)d_in[8];
    const float* outW  = (const float*)d_in[9];
    const float* outb  = (const float*)d_in[10];
    const float* eWih  = (const float*)d_in[11];
    const float* eWhh  = (const float*)d_in[12];
    const float* ebih  = (const float*)d_in[13];
    const float* ebhh  = (const float*)d_in[14];
    const float* dWih  = (const float*)d_in[15];
    const float* dWhh  = (const float*)d_in[16];
    const float* dbih  = (const float*)d_in[17];
    const float* dbhh  = (const float*)d_in[18];
    float* out = (float*)d_out;

    // workspace carving (256B aligned)
    char* p = (char*)d_ws;
    auto take = [&](size_t bytes) -> char* {
        char* q = p;
        p += (bytes + 255) & ~(size_t)255;
        return q;
    };
    unsigned short* Wc_e0 = (unsigned short*)take((size_t)G4H * KXH * 2);
    unsigned short* Wc_e1 = (unsigned short*)take((size_t)G4H * KXH * 2);
    unsigned short* Wc_d0 = (unsigned short*)take((size_t)G4H * KXH * 2);
    unsigned short* Wc_d1 = (unsigned short*)take((size_t)G4H * KXH * 2);
    unsigned short* embWp = (unsigned short*)take((size_t)HN * KIN * 2);
    unsigned short* outWp = (unsigned short*)take((size_t)FNP * HN * 2);
    unsigned short* xh0[2];
    unsigned short* xh1[2];
    for (int i = 0; i < 2; ++i) xh0[i] = (unsigned short*)take((size_t)BN * KXH * 2);
    for (int i = 0; i < 2; ++i) xh1[i] = (unsigned short*)take((size_t)BN * KXH * 2);
    float* c0 = (float*)take((size_t)BN * HN * 4);
    float* c1 = (float*)take((size_t)BN * HN * 4);
    unsigned short* genin = (unsigned short*)take((size_t)BN * KIN * 2);

    // --- setup: pack weights to bf16, build static gen_in, zero states ---
    pack_lstm_w_kernel<<<8192, 256, 0, stream>>>(eWih, eWhh, Wc_e0);
    pack_lstm_w_kernel<<<8192, 256, 0, stream>>>(eWih + (size_t)G4H * HN,
                                                 eWhh + (size_t)G4H * HN, Wc_e1);
    pack_lstm_w_kernel<<<8192, 256, 0, stream>>>(dWih, dWhh, Wc_d0);
    pack_lstm_w_kernel<<<8192, 256, 0, stream>>>(dWih + (size_t)G4H * HN,
                                                 dWhh + (size_t)G4H * HN, Wc_d1);
    pack_emb_w_kernel<<<(HN * KIN + 255) / 256, 256, 0, stream>>>(embW, embWp);
    pack_out_w_kernel<<<(FNP * HN) / 256, 256, 0, stream>>>(outW, outWp);
    build_genin_kernel<<<(BN * KIN + 255) / 256, 256, 0, stream>>>(label, cemb, genin);
    for (int i = 0; i < 2; ++i) {
        zero_u32_kernel<<<1024, 256, 0, stream>>>((unsigned int*)xh0[i], BN * KXH / 2);
        zero_u32_kernel<<<1024, 256, 0, stream>>>((unsigned int*)xh1[i], BN * KXH / 2);
    }
    zero_u32_kernel<<<1024, 256, 0, stream>>>((unsigned int*)c0, BN * HN);
    zero_u32_kernel<<<1024, 256, 0, stream>>>((unsigned int*)c1, BN * HN);

    // --- encoder: 120 steps, ping-pong on parity of global step s ---
    for (int t = 0; t < TT; ++t) {
        const int pgt = t & 1, nx = 1 - pgt;
        traj_embed_kernel<<<1024, 256, 0, stream>>>(traj, encW, encb, prelu, t, xh0[pgt]);
        lstm_step_kernel<<<128, 128, 0, stream>>>(xh0[pgt], Wc_e0, ebih, ebhh,
                                                  c0, xh0[nx], xh1[pgt]);
        lstm_step_kernel<<<128, 128, 0, stream>>>(xh1[pgt], Wc_e1, ebih + G4H, ebhh + G4H,
                                                  c1, xh1[nx], nullptr);
    }

    // --- decoder: 120 steps (s = 120+t, parity = t&1 since 120 is even) ---
    for (int t = 0; t < TT; ++t) {
        const int pgt = t & 1, nx = 1 - pgt;
        set_tfrac_kernel<<<2, 256, 0, stream>>>(genin, (float)(t + 1) / (float)TT);
        emb_gemm_kernel<<<128, 128, 0, stream>>>(genin, embWp, embb, prelu, xh0[pgt]);
        lstm_step_kernel<<<128, 128, 0, stream>>>(xh0[pgt], Wc_d0, dbih, dbhh,
                                                  c0, xh0[nx], xh1[pgt]);
        lstm_step_kernel<<<128, 128, 0, stream>>>(xh1[pgt], Wc_d1, dbih + G4H, dbhh + G4H,
                                                  c1, xh1[nx], nullptr);
        // new h of layer1 lives in xh1[nx][:,512:]
        out_gemm_kernel<<<20, 128, 0, stream>>>(xh1[nx] + HN, outWp, outb, t, out, genin);
    }
}
